// DecoderRNN_59219009077297
// MI455X (gfx1250) — matrix operand order repaired and checked by reading, not compiled, not run on hardware
//
#include <hip/hip_runtime.h>
#include <hip/hip_bf16.h>

// ---------------- problem constants ----------------
#define B_     32
#define RNN_   512
#define EMB_   512
#define VOCAB_ 50257
#define VPAD_  50272   // next multiple of 32 (50272/32 = 1571)
#define LEN_   64
#define NG_    1536    // 3*RNN
#define KEEP_  0.9f
#define EPSR_  1e-20f
#define TDM_KC 64      // K-chunk staged per tensor_load_to_lds

typedef __attribute__((ext_vector_type(16))) __bf16   v16bf;
typedef __attribute__((ext_vector_type(8)))  float    v8f;
typedef __attribute__((ext_vector_type(4)))  unsigned v4u;
typedef __attribute__((ext_vector_type(8)))  unsigned v8u;

// ---------------- helpers ----------------
__device__ __forceinline__ __bf16 to_bf16(float f) {
  unsigned u = __builtin_bit_cast(unsigned, f);
  unsigned r = u + 0x7FFFu + ((u >> 16) & 1u);   // round-to-nearest-even
  unsigned short s = (unsigned short)(r >> 16);
  return __builtin_bit_cast(__bf16, s);
}

__device__ __forceinline__ unsigned hash32(unsigned x) {
  x ^= x >> 16; x *= 0x7feb352du;
  x ^= x >> 15; x *= 0x846ca68bu;
  x ^= x >> 16; return x;
}

__device__ __forceinline__ float uniform01(unsigned h) {
  return (float)(h >> 8) * (1.0f / 16777216.0f);
}

__device__ __forceinline__ float gumbel_noise(unsigned seed, int b, int v) {
  unsigned h = hash32(seed ^ (unsigned)(b * 0x9E3779B9u) ^ (unsigned)(v * 0x85EBCA6Bu));
  float U = uniform01(h);
  return -__logf(-__logf(U + EPSR_) + EPSR_);
}

// ---------------- TDM: stage a [rows x kc] bf16 tile (row stride = strideElems)
// from global into LDS at byte offset ldsOff. D# built per ISA 8.3/8.4:
//   group0: count=1 | lds_addr | global_addr | type=2
//   group1: data_size=2B, tensor dims large (no OOB clamp), tile = kc x rows,
//           tensor_dim0_stride = strideElems
__device__ __forceinline__ void tdm_load_tile(unsigned ldsOff, const __bf16* gptr,
                                              int rows, int kc, int strideElems) {
  unsigned long long ga = (unsigned long long)(size_t)gptr;
  unsigned g0w0 = 1u;                                    // count = 1 valid D#
  unsigned g0w1 = ldsOff;                                // LDS byte address
  unsigned g0w2 = (unsigned)(ga & 0xFFFFFFFFu);          // global_addr[31:0]
  unsigned g0w3 = (unsigned)((ga >> 32) & 0x1FFFFFFu)    // global_addr[56:32]
                | (2u << 30);                            // type = 2 ("image")
  const unsigned td0 = 1u << 20;                         // tensor_dim0 (large)
  const unsigned td1 = 1u << 16;                         // tensor_dim1 (large)
  unsigned g1w0 = 1u << 16;                              // data_size = 1 (2 bytes)
  unsigned g1w1 = (td0 & 0xFFFFu) << 16;                 // tensor_dim0[15:0]
  unsigned g1w2 = (td0 >> 16) | ((td1 & 0xFFFFu) << 16); // dim0[31:16] | dim1[15:0]
  unsigned g1w3 = (td1 >> 16) | ((unsigned)kc << 16);    // dim1[31:16] | tile_dim0
  unsigned g1w4 = (unsigned)rows;                        // tile_dim1 (tile_dim2 = 0)
  unsigned g1w5 = (unsigned)strideElems;                 // tensor_dim0_stride[31:0]
  unsigned g1w6 = 0u;
  unsigned g1w7 = 0u;
  v4u g0 = { (unsigned)__builtin_amdgcn_readfirstlane((int)g0w0),
             (unsigned)__builtin_amdgcn_readfirstlane((int)g0w1),
             (unsigned)__builtin_amdgcn_readfirstlane((int)g0w2),
             (unsigned)__builtin_amdgcn_readfirstlane((int)g0w3) };
  v8u g1 = { (unsigned)__builtin_amdgcn_readfirstlane((int)g1w0),
             (unsigned)__builtin_amdgcn_readfirstlane((int)g1w1),
             (unsigned)__builtin_amdgcn_readfirstlane((int)g1w2),
             (unsigned)__builtin_amdgcn_readfirstlane((int)g1w3),
             (unsigned)__builtin_amdgcn_readfirstlane((int)g1w4),
             (unsigned)__builtin_amdgcn_readfirstlane((int)g1w5),
             (unsigned)__builtin_amdgcn_readfirstlane((int)g1w6),
             (unsigned)__builtin_amdgcn_readfirstlane((int)g1w7) };
  asm volatile("tensor_load_to_lds %0, %1" :: "s"(g0), "s"(g1) : "memory");
}

// ---------------- weight prep: fp32 [K x N] -> bf16 [Npad x K] ----
__global__ void k_transpose_padN(const float* __restrict__ src, int K, int N,
                                 int Npad, __bf16* __restrict__ dst) {
  size_t tid = (size_t)blockIdx.x * blockDim.x + threadIdx.x;
  size_t total = (size_t)Npad * K;
  if (tid >= total) return;
  int n = (int)(tid / K);
  int k = (int)(tid % K);
  float v = (n < N) ? src[(size_t)k * N + n] : 0.0f;
  dst[tid] = to_bf16(v);
}

// fp32 [K x N] -> bf16 [N x Kpad] (pad K dimension with zeros)
__global__ void k_transpose_padK(const float* __restrict__ src, int K, int N,
                                 int Kpad, __bf16* __restrict__ dst) {
  size_t tid = (size_t)blockIdx.x * blockDim.x + threadIdx.x;
  size_t total = (size_t)N * Kpad;
  if (tid >= total) return;
  int n = (int)(tid / Kpad);
  int k = (int)(tid % Kpad);
  float v = (k < K) ? src[(size_t)k * N + n] : 0.0f;
  dst[tid] = to_bf16(v);
}

// ---------------- A-fragment pre-pack --------------------------------------
// Pack activation A[32 x K] fp32 into a 16-bit A 16x32 WMMA fragment
// layout (contiguous k = 16*half + e): Apack[((mTile*(K/32)+kstep)*32 + lane)*16 + e], so the
// GEMM inner loop does a single contiguous 32B load per fragment.
__global__ void k_pack_a(const float* __restrict__ A, int lda, int K,
                         __bf16* __restrict__ Apack) {
  int idx = blockIdx.x * blockDim.x + threadIdx.x;
  if (idx >= B_ * K) return;
  int e    = idx & 15;
  int lane = (idx >> 4) & 31;
  int grp  = idx >> 9;                 // mTile*(K/32) + kstep
  int ks   = K >> 5;
  int s    = grp % ks;
  int m    = grp / ks;
  int half = lane >> 4, l16 = lane & 15;
  int row  = m * 16 + l16;
  int kk   = s * 32 + half * 16 + e;
  Apack[idx] = to_bf16(A[(size_t)row * lda + kk]);
}

// ---------------- WMMA GEMM (global B): C[32 x N] = A*B (+bias) ------------
// Apack: pre-packed fragments; Bt[n*ldb + k] bf16 (pre-transposed).
// One wave per 16x16 tile; optional split-K via atomicAdd.
__global__ void k_wmma_gemm(const __bf16* __restrict__ Apack,
                            const __bf16* __restrict__ Bt, int ldb,
                            const float* __restrict__ bias,
                            float* __restrict__ C, long long ldc,
                            int N, int K, int nTiles, int kSplits) {
  const int wavesPerBlock = blockDim.x >> 5;
  const int wave = blockIdx.x * wavesPerBlock + (threadIdx.x >> 5);
  const int totalWork = 2 * nTiles * kSplits;
  if (wave >= totalWork) return;        // wave-uniform: EXEC stays all-ones

  const int mTile = wave & 1;
  const int rest  = wave >> 1;
  const int nTile = rest % nTiles;
  const int kIdx  = rest / nTiles;

  const int ksteps   = K >> 5;
  const int perSplit = ((ksteps + kSplits - 1) / kSplits) << 5;
  const int k0 = kIdx * perSplit;
  const int k1 = (k0 + perSplit < K) ? (k0 + perSplit) : K;

  const int lane = threadIdx.x & 31;
  const int half = lane >> 4;
  const int l16  = lane & 15;

  const __bf16* afrag = Apack + ((size_t)mTile * ksteps * 32 + lane) * 16;
  const __bf16* brow  = Bt + (size_t)(nTile * 16 + l16) * ldb;

  v8f acc = {};
#pragma unroll 4
  for (int k = k0; k < k1; k += 32) {
    __builtin_prefetch((const char*)(brow + k + 128), 0, 1);   // global_prefetch_b8
    const v16bf af  = *(const v16bf*)(afrag + (size_t)(k >> 5) * 32 * 16);
    const v16bf bfv = *(const v16bf*)(brow + k + half * 16);
    acc = __builtin_amdgcn_wmma_f32_16x16x32_bf16(
        false, af, false, bfv, (short)0, acc, false, false);
  }

  const int col = nTile * 16 + l16;
  if (col < N) {
    const float bv = (bias && kIdx == 0) ? bias[col] : 0.0f;
    if (kSplits == 1) {
#pragma unroll
      for (int j = 0; j < 8; ++j)
        C[(size_t)(mTile * 16 + j + half * 8) * ldc + col] = acc[j] + bv;
    } else {
#pragma unroll
      for (int j = 0; j < 8; ++j)
        atomicAdd(&C[(size_t)(mTile * 16 + j + half * 8) * ldc + col], acc[j] + bv);
    }
  }
}

// ---------------- WMMA GEMM with TDM-staged B (logits GEMM) ----------------
// One wave per N-tile, computing BOTH 16-row M halves (B staged once, feeds 2
// WMMAs per k-step). B tile is DMA'd into LDS in TDM_KC-wide chunks by the
// Tensor Data Mover, double-buffered, synchronized with s_wait_tensorcnt.
__global__ void k_wmma_gemm_tdm(const __bf16* __restrict__ Apack,
                                const __bf16* __restrict__ Bt, int ldb,
                                const float* __restrict__ bias,
                                float* __restrict__ C, long long ldc,
                                int N, int K, int nTiles) {
  __shared__ __bf16 ldsB[8 * 2 * 16 * TDM_KC];   // 8 waves x double buffer = 32KB
  const int wavesPerBlock = blockDim.x >> 5;
  const int waveInBlk = threadIdx.x >> 5;
  const int wave = blockIdx.x * wavesPerBlock + waveInBlk;
  if (wave >= nTiles) return;                    // wave-uniform

  const int nTile = wave;
  const int lane = threadIdx.x & 31;
  const int half = lane >> 4;
  const int l16  = lane & 15;
  const int ksteps = K >> 5;

  __bf16* myLds = ldsB + (size_t)waveInBlk * (2 * 16 * TDM_KC);
  const unsigned ldsBase = (unsigned)(size_t)myLds;          // LDS byte offset
  const __bf16* gTile = Bt + (size_t)nTile * 16 * ldb;       // 16 rows of B^T
  const int nChunks = K / TDM_KC;

  const __bf16* afrag0 = Apack + ((size_t)0 * ksteps * 32 + lane) * 16;
  const __bf16* afrag1 = Apack + ((size_t)1 * ksteps * 32 + lane) * 16;

  // prologue: stage chunk 0 into buffer 0
  tdm_load_tile(ldsBase, gTile, 16, TDM_KC, ldb);

  v8f acc0 = {}, acc1 = {};
  for (int c = 0; c < nChunks; ++c) {
    if (c + 1 < nChunks) {
      tdm_load_tile(ldsBase + (unsigned)(((c + 1) & 1) * 16 * TDM_KC * 2),
                    gTile + (size_t)(c + 1) * TDM_KC, 16, TDM_KC, ldb);
      __builtin_amdgcn_s_wait_tensorcnt((short)1);   // chunk c resident
    } else {
      __builtin_amdgcn_s_wait_tensorcnt((short)0);   // last chunk resident
    }
    const __bf16* bbuf = myLds + (size_t)(c & 1) * 16 * TDM_KC;
#pragma unroll
    for (int kk = 0; kk < TDM_KC; kk += 32) {
      const int s = (c * TDM_KC + kk) >> 5;
      const v16bf a0 = *(const v16bf*)(afrag0 + (size_t)s * 32 * 16);
      const v16bf a1 = *(const v16bf*)(afrag1 + (size_t)s * 32 * 16);
      const v16bf bv = *(const v16bf*)(bbuf + (size_t)l16 * TDM_KC + kk + 16 * half);
      acc0 = __builtin_amdgcn_wmma_f32_16x16x32_bf16(
          false, a0, false, bv, (short)0, acc0, false, false);
      acc1 = __builtin_amdgcn_wmma_f32_16x16x32_bf16(
          false, a1, false, bv, (short)0, acc1, false, false);
    }
  }

  const int col = nTile * 16 + l16;
  if (col < N) {
    const float bv = bias ? bias[col] : 0.0f;
#pragma unroll
    for (int j = 0; j < 8; ++j) {
      C[(size_t)(j + half * 8) * ldc + col]      = acc0[j] + bv;
      C[(size_t)(16 + j + half * 8) * ldc + col] = acc1[j] + bv;
    }
  }
}

// ---------------- GRU gate combine ----------------
__global__ void k_gru_combine(const float* __restrict__ Gx,
                              const float* __restrict__ Gh,
                              const float* __restrict__ hcur,
                              float* __restrict__ hnew) {
  int idx = blockIdx.x * blockDim.x + threadIdx.x;
  if (idx >= B_ * RNN_) return;
  int b = idx >> 9;
  int i = idx & 511;
  const float* gx = Gx + (size_t)b * NG_;
  const float* gh = Gh + (size_t)b * NG_;
  float z = 1.0f / (1.0f + __expf(-(gx[i] + gh[i])));
  float r = 1.0f / (1.0f + __expf(-(gx[RNN_ + i] + gh[RNN_ + i])));
  float n = tanhf(gx[2 * RNN_ + i] + r * gh[2 * RNN_ + i]);
  float h = hcur[idx];
  hnew[idx] = z * h + (1.0f - z) * n;
}

// ---------------- dropout (inverted scaling) ----------------
__global__ void k_dropout(const float* __restrict__ hnew,
                          float* __restrict__ outd, unsigned seed) {
  int idx = blockIdx.x * blockDim.x + threadIdx.x;
  if (idx >= B_ * RNN_) return;
  float u = uniform01(hash32(seed ^ (unsigned)(idx * 0x9E3779B9u)));
  outd[idx] = (u < KEEP_) ? hnew[idx] * (1.0f / KEEP_) : 0.0f;
}

// ---------------- store pre-step h into h_seq output ----------------
__global__ void k_store_h(const float* __restrict__ h, float* __restrict__ out,
                          int t) {
  int idx = blockIdx.x * blockDim.x + threadIdx.x;
  if (idx >= B_ * RNN_) return;
  int b = idx >> 9;
  int i = idx & 511;
  out[(size_t)b * LEN_ * RNN_ + (size_t)t * RNN_ + i] = h[idx];
}

// ---------------- gumbel softmax -> prob (one block per batch row) --------
__global__ void k_softmax_sample(const float* __restrict__ logits,
                                 long long rowStride,
                                 const float* __restrict__ gammaPtr,
                                 float* __restrict__ prob, unsigned seed) {
  __shared__ float red[256];
  const int b = blockIdx.x;
  const int tid = threadIdx.x;
  const float* lrow = logits + (size_t)b * rowStride;
  const float inv_g = 1.0f / gammaPtr[0];

  float m = -3.4e38f;
  for (int v = tid; v < VOCAB_; v += 256) {
    float s = (lrow[v] + gumbel_noise(seed, b, v)) * inv_g;
    m = fmaxf(m, s);
  }
  red[tid] = m; __syncthreads();
  for (int s = 128; s > 0; s >>= 1) {
    if (tid < s) red[tid] = fmaxf(red[tid], red[tid + s]);
    __syncthreads();
  }
  m = red[0]; __syncthreads();

  float sum = 0.0f;
  for (int v = tid; v < VOCAB_; v += 256) {
    float s = (lrow[v] + gumbel_noise(seed, b, v)) * inv_g;
    sum += __expf(s - m);
  }
  red[tid] = sum; __syncthreads();
  for (int s = 128; s > 0; s >>= 1) {
    if (tid < s) red[tid] += red[tid + s];
    __syncthreads();
  }
  sum = red[0]; __syncthreads();
  const float inv = 1.0f / sum;

  for (int v = tid; v < VPAD_; v += 256) {
    float p = 0.0f;
    if (v < VOCAB_) {
      float s = (lrow[v] + gumbel_noise(seed, b, v)) * inv_g;
      p = __expf(s - m) * inv;
    }
    prob[(size_t)b * VPAD_ + v] = p;
  }
}

// ---------------- host orchestration ----------------
extern "C" void kernel_launch(void* const* d_in, const int* in_sizes, int n_in,
                              void* d_out, int out_size, void* d_ws, size_t ws_size,
                              hipStream_t stream) {
  (void)in_sizes; (void)n_in; (void)out_size; (void)ws_size;
  const float* in_h   = (const float*)d_in[0];
  const float* in_x   = (const float*)d_in[1];
  const float* in_emb = (const float*)d_in[2];
  const float* in_Wx  = (const float*)d_in[3];
  const float* in_Wh  = (const float*)d_in[4];
  const float* in_bg  = (const float*)d_in[5];
  const float* in_Wd  = (const float*)d_in[6];
  const float* in_bd  = (const float*)d_in[7];
  const float* in_gam = (const float*)d_in[8];

  float* out = (float*)d_out;
  const size_t HSEQ = (size_t)B_ * LEN_ * RNN_;
  float* out_logits = out + HSEQ;                        // [B, LEN, VOCAB]

  // -------- workspace carve (256B aligned) --------
  size_t off = 0;
  auto carve = [&](size_t bytes) -> char* {
    char* p = (char*)d_ws + off;
    off = (off + bytes + 255) & ~(size_t)255;
    return p;
  };
  __bf16* WxT   = (__bf16*)carve((size_t)NG_ * EMB_ * 2);
  __bf16* WhT   = (__bf16*)carve((size_t)NG_ * RNN_ * 2);
  __bf16* WdT   = (__bf16*)carve((size_t)VPAD_ * RNN_ * 2);
  __bf16* embT  = (__bf16*)carve((size_t)EMB_ * VPAD_ * 2);
  __bf16* ApkX  = (__bf16*)carve((size_t)B_ * EMB_ * 2);
  __bf16* ApkH  = (__bf16*)carve((size_t)B_ * RNN_ * 2);
  __bf16* ApkD  = (__bf16*)carve((size_t)B_ * RNN_ * 2);
  __bf16* ApkP  = (__bf16*)carve((size_t)B_ * VPAD_ * 2);
  float* Gx     = (float*)carve((size_t)B_ * NG_ * 4);
  float* Gh     = (float*)carve((size_t)B_ * NG_ * 4);
  float* drop   = (float*)carve((size_t)B_ * RNN_ * 4);
  float* prob   = (float*)carve((size_t)B_ * VPAD_ * 4);
  float* hbuf[2] = { (float*)carve((size_t)B_ * RNN_ * 4),
                     (float*)carve((size_t)B_ * RNN_ * 4) };
  float* xbuf[2] = { (float*)carve((size_t)B_ * EMB_ * 4),
                     (float*)carve((size_t)B_ * EMB_ * 4) };

  const int T256 = 256;
  auto nblk = [](size_t n, int t) { return (int)((n + t - 1) / t); };

  // -------- per-launch weight conversion / transpose --------
  k_transpose_padN<<<nblk((size_t)NG_ * EMB_, T256), T256, 0, stream>>>(
      in_Wx, EMB_, NG_, NG_, WxT);
  k_transpose_padN<<<nblk((size_t)NG_ * RNN_, T256), T256, 0, stream>>>(
      in_Wh, RNN_, NG_, NG_, WhT);
  k_transpose_padN<<<nblk((size_t)VPAD_ * RNN_, T256), T256, 0, stream>>>(
      in_Wd, RNN_, VOCAB_, VPAD_, WdT);
  k_transpose_padK<<<nblk((size_t)EMB_ * VPAD_, T256), T256, 0, stream>>>(
      in_emb, VOCAB_, EMB_, VPAD_, embT);

  // -------- initial state --------
  hipMemcpyAsync(hbuf[0], in_h, (size_t)B_ * RNN_ * 4,
                 hipMemcpyDeviceToDevice, stream);
  hipMemcpyAsync(xbuf[0], in_x, (size_t)B_ * EMB_ * 4,
                 hipMemcpyDeviceToDevice, stream);

  const int nTilesG  = NG_ / 16;                    // 96
  const int blocksG  = (2 * nTilesG + 7) / 8;       // 24
  const int nTilesV  = (VOCAB_ + 15) / 16;          // 3142
  const int blocksV  = (nTilesV + 7) / 8;           // 393 (one wave per N tile)
  const int nTilesE  = EMB_ / 16;                   // 32
  const int kSplitsE = 16;
  const int blocksE  = (2 * nTilesE * kSplitsE + 7) / 8;

  for (int t = 0; t < LEN_; ++t) {
    float* hc = hbuf[t & 1];
    float* hn = hbuf[(t + 1) & 1];
    float* xc = xbuf[t & 1];
    float* xn = xbuf[(t + 1) & 1];

    // h_seq[:, t, :] = h_t (pre-step)
    k_store_h<<<nblk(B_ * RNN_, T256), T256, 0, stream>>>(hc, out, t);

    // Gx = x @ Wx + b_gru ; Gh = h @ Wh
    k_pack_a<<<nblk(B_ * EMB_, T256), T256, 0, stream>>>(xc, EMB_, EMB_, ApkX);
    k_wmma_gemm<<<blocksG, 256, 0, stream>>>(ApkX, WxT, EMB_, in_bg,
                                             Gx, NG_, NG_, EMB_, nTilesG, 1);
    k_pack_a<<<nblk(B_ * RNN_, T256), T256, 0, stream>>>(hc, RNN_, RNN_, ApkH);
    k_wmma_gemm<<<blocksG, 256, 0, stream>>>(ApkH, WhT, RNN_, nullptr,
                                             Gh, NG_, NG_, RNN_, nTilesG, 1);

    // h_new = GRU(Gx, Gh, h)
    k_gru_combine<<<nblk(B_ * RNN_, T256), T256, 0, stream>>>(Gx, Gh, hc, hn);

    // out = dropout(h_new)
    k_dropout<<<nblk(B_ * RNN_, T256), T256, 0, stream>>>(
        hn, drop, 0xD00D0000u + (unsigned)t);

    // logits[:, t, :] = out @ Wd + bd  (TDM-staged B, straight into d_out)
    k_pack_a<<<nblk(B_ * RNN_, T256), T256, 0, stream>>>(drop, RNN_, RNN_, ApkD);
    k_wmma_gemm_tdm<<<blocksV, 256, 0, stream>>>(
        ApkD, WdT, RNN_, in_bd,
        out_logits + (size_t)t * VOCAB_, (long long)LEN_ * VOCAB_,
        VOCAB_, RNN_, nTilesV);

    // prob = gumbel_softmax(logits / gamma)
    k_softmax_sample<<<B_, 256, 0, stream>>>(
        out_logits + (size_t)t * VOCAB_, (long long)LEN_ * VOCAB_,
        in_gam, prob, 0x6B450000u + (unsigned)t);

    // x_new = prob @ embedding  (split-K, atomic accumulate)
    k_pack_a<<<nblk(B_ * VPAD_, T256), T256, 0, stream>>>(prob, VPAD_, VPAD_, ApkP);
    hipMemsetAsync(xn, 0, (size_t)B_ * EMB_ * 4, stream);
    k_wmma_gemm<<<blocksE, 256, 0, stream>>>(
        ApkP, embT, VPAD_, nullptr,
        xn, EMB_, EMB_, VPAD_, nTilesE, kSplitsE);
  }
}
